// SparseMoeFFN_18640158065027
// MI455X (gfx1250) — matrix operand order repaired and checked by reading, not compiled, not run on hardware
//
#include <hip/hip_runtime.h>

#define TOP_K 2
#define NUM_EXPERTS 8
#define HIDDEN 1024
#define INTER 2048

typedef __attribute__((ext_vector_type(16))) __bf16 v16bf;
typedef __attribute__((ext_vector_type(8)))  float  v8f;

// ---------- helpers ----------

__device__ __forceinline__ __bf16 f2bf(float f) {
    return (__bf16)f;   // hardware RNE f32->bf16 cvt
}

struct alignas(16) BF8 { __bf16 v[8]; };

// A-matrix fragment (16x32 bf16) from an LDS row (row-major, bf16):
// lane L holds row M=L&15; element j <-> K = (L>>4)*16 + j
__device__ __forceinline__ v16bf load_a_frag(const __bf16* row, int kb, int hi8) {
    const BF8* p0 = reinterpret_cast<const BF8*>(row + kb + hi8 * 2);
    const BF8* p1 = p0 + 1;
    v16bf a;
#pragma unroll
    for (int j = 0; j < 8; ++j) { a[j] = p0->v[j]; a[8 + j] = p1->v[j]; }
    return a;
}

// B-matrix fragment (32x16 bf16) from pre-transposed bf16 weights WT[N][K]:
// lane L holds col N=L&15 (a row of WT); element j <-> K = kb + (L>>4)*16 + j
// -> 32 contiguous bytes per lane: two aligned b128 loads.
__device__ __forceinline__ v16bf load_b_frag(const __bf16* wt_row, int kb, int hi16) {
    const BF8* p0 = reinterpret_cast<const BF8*>(wt_row + kb + hi16);
    const BF8* p1 = p0 + 1;
    v16bf b;
#pragma unroll
    for (int j = 0; j < 8; ++j) { b[j] = p0->v[j]; b[8 + j] = p1->v[j]; }
    return b;
}

__device__ __forceinline__ v8f wmma_bf16(v16bf a, v16bf b, v8f c) {
    return __builtin_amdgcn_wmma_f32_16x16x32_bf16(false, a, false, b, (short)0, c, false, false);
}

// ---------- kernel 1: zero output + counters ----------

__global__ void moe_init_kernel(float* __restrict__ out, int* __restrict__ cnt, int n4) {
    int i = blockIdx.x * blockDim.x + threadIdx.x;
    if (i < n4) reinterpret_cast<float4*>(out)[i] = float4{0.f, 0.f, 0.f, 0.f};
    if (i < NUM_EXPERTS) cnt[i] = 0;
}

// ---------- kernel 2: x fp32 -> bf16 ----------

struct alignas(8) BF4 { __bf16 v[4]; };

__global__ void moe_cvt_kernel(const float* __restrict__ x, __bf16* __restrict__ xb, int n4) {
    int i = blockIdx.x * blockDim.x + threadIdx.x;
    if (i >= n4) return;
    float4 f = reinterpret_cast<const float4*>(x)[i];
    BF4 o;
    o.v[0] = f2bf(f.x); o.v[1] = f2bf(f.y); o.v[2] = f2bf(f.z); o.v[3] = f2bf(f.w);
    reinterpret_cast<BF4*>(xb)[i] = o;
}

// ---------- kernel 3: weight transpose + convert: W[R][C] f32 -> WT[C][R] bf16 ----------
// grid: (C/32, R/32, experts); block: 256

__global__ void moe_wt_kernel(const float* __restrict__ W, __bf16* __restrict__ WT,
                              int R, int C) {
    __shared__ __bf16 tile[32][33];
    const size_t esz = (size_t)R * C;
    const float* Wi  = W  + esz * blockIdx.z;
    __bf16* WTo      = WT + esz * blockIdx.z;

    int tx = threadIdx.x & 31, ty = threadIdx.x >> 5;     // 32 x 8
    int c0 = blockIdx.x * 32, r0 = blockIdx.y * 32;
#pragma unroll
    for (int i = 0; i < 4; ++i) {
        int r = ty + i * 8;
        tile[r][tx] = f2bf(Wi[(size_t)(r0 + r) * C + c0 + tx]);
    }
    __syncthreads();
#pragma unroll
    for (int i = 0; i < 4; ++i) {
        int c = ty + i * 8;
        WTo[(size_t)(c0 + c) * R + r0 + tx] = tile[tx][c];
    }
}

// ---------- kernel 4: router (one wave32 per token) ----------

__global__ void moe_router_kernel(const float* __restrict__ x, const float* __restrict__ rw,
                                  int* __restrict__ cnt, int* __restrict__ idxb,
                                  float* __restrict__ wtb, int T) {
    int tid  = threadIdx.x;
    int lane = tid & 31;
    int tok  = blockIdx.x * (blockDim.x >> 5) + (tid >> 5);
    if (tok >= T) return;

    const float* xr = x + (size_t)tok * HIDDEN;
    float logit[NUM_EXPERTS];
#pragma unroll
    for (int e = 0; e < NUM_EXPERTS; ++e) {
        const float* w = rw + e * HIDDEN;
        float acc = 0.f;
        for (int h = lane; h < HIDDEN; h += 32) acc = fmaf(xr[h], w[h], acc);
#pragma unroll
        for (int o = 16; o >= 1; o >>= 1) acc += __shfl_xor(acc, o, 32);
        logit[e] = acc;
    }
    if (lane == 0) {
        int i0 = 0; float m0 = logit[0];
#pragma unroll
        for (int e = 1; e < NUM_EXPERTS; ++e)
            if (logit[e] > m0) { m0 = logit[e]; i0 = e; }
        int i1 = -1; float m1 = -3.4e38f;
#pragma unroll
        for (int e = 0; e < NUM_EXPERTS; ++e)
            if (e != i0 && logit[e] > m1) { m1 = logit[e]; i1 = e; }
        float e1  = __expf(m1 - m0);
        float inv = __builtin_amdgcn_rcpf(1.f + e1);
        int s0 = atomicAdd(&cnt[i0], 1);
        idxb[i0 * T + s0] = tok; wtb[i0 * T + s0] = inv;
        int s1 = atomicAdd(&cnt[i1], 1);
        idxb[i1 * T + s1] = tok; wtb[i1 * T + s1] = e1 * inv;
    }
}

// ---------- kernel 5: fused expert FFN, WMMA bf16, MTILE=32 ----------

#define MTILE   32
#define XPITCH  (HIDDEN + 8)   // +16B pad: 16 lane-rows spread over all 64 banks
#define HPITCH  (INTER + 8)
#define XS_BYTES  (MTILE * XPITCH * 2)
#define HS_BYTES  (MTILE * HPITCH * 2)
#define SMEM_BYTES (XS_BYTES + HS_BYTES + 2 * MTILE * 4)

__global__ void __launch_bounds__(256)
moe_ffn_kernel(const __bf16* __restrict__ xb,
               const __bf16* __restrict__ gwt, const __bf16* __restrict__ uwt,
               const __bf16* __restrict__ dwt,
               const int* __restrict__ cnt, const int* __restrict__ idxb,
               const float* __restrict__ wtb,
               float* __restrict__ out, int T) {
    extern __shared__ char smem[];
    __bf16* Xs  = reinterpret_cast<__bf16*>(smem);                       // [32][XPITCH]
    __bf16* Hs  = reinterpret_cast<__bf16*>(smem + XS_BYTES);            // [32][HPITCH]
    int*   toks = reinterpret_cast<int*>(smem + XS_BYTES + HS_BYTES);    // [32]
    float* twgt = reinterpret_cast<float*>(toks + MTILE);                // [32]

    const int e  = blockIdx.y;
    const int n  = cnt[e];
    const int r0 = blockIdx.x * MTILE;
    if (r0 >= n) return;

    const int tid  = threadIdx.x;          // 256 threads = 8 waves
    const int wave = tid >> 5;
    const int lane = tid & 31;
    const int hi8  = (lane >> 4) * 8;
    const int hi16 = (lane >> 4) * 16;
    const int nl   = lane & 15;

    if (tid < MTILE) {
        int r = r0 + tid;
        toks[tid] = (r < n) ? idxb[e * T + r] : 0;
        twgt[tid] = (r < n) ? wtb[e * T + r] : 0.f;   // pad rows -> weight 0 (inert)
    }
    __syncthreads();

    // gather 32 token rows of bf16 x into LDS (8 threads per row, 256B each)
    {
        int row = tid >> 3, seg = tid & 7;
        const float4* src = reinterpret_cast<const float4*>(xb + (size_t)toks[row] * HIDDEN + seg * 128);
        float4* dst = reinterpret_cast<float4*>(Xs + row * XPITCH + seg * 128);
#pragma unroll
        for (int j = 0; j < 16; ++j) dst[j] = src[j];
    }
    __syncthreads();

    const __bf16* Gt = gwt + (size_t)e * HIDDEN * INTER;   // [INTER][HIDDEN]
    const __bf16* Ut = uwt + (size_t)e * HIDDEN * INTER;   // [INTER][HIDDEN]
    const __bf16* Dt = dwt + (size_t)e * INTER * HIDDEN;   // [HIDDEN][INTER]
    const __bf16* arow0 = Xs + nl * XPITCH;
    const __bf16* arow1 = Xs + (16 + nl) * XPITCH;

    // ---- phase A: H[32][2048] = silu(X@G) * (X@U); each wave owns 256 INTER cols ----
    for (int c = 0; c < 16; ++c) {
        const int n0 = wave * 256 + c * 16;
        const __bf16* grow = Gt + (size_t)(n0 + nl) * HIDDEN;
        const __bf16* urow = Ut + (size_t)(n0 + nl) * HIDDEN;
        v8f g0, g1, u0, u1;
#pragma unroll
        for (int i = 0; i < 8; ++i) { g0[i] = 0.f; g1[i] = 0.f; u0[i] = 0.f; u1[i] = 0.f; }
        for (int kb = 0; kb < HIDDEN; kb += 32) {
            v16bf a0 = load_a_frag(arow0, kb, hi8);
            v16bf a1 = load_a_frag(arow1, kb, hi8);
            v16bf bg = load_b_frag(grow, kb, hi16);
            v16bf bu = load_b_frag(urow, kb, hi16);
            g0 = wmma_bf16(a0, bg, g0);
            g1 = wmma_bf16(a1, bg, g1);
            u0 = wmma_bf16(a0, bu, u0);
            u1 = wmma_bf16(a1, bu, u1);
        }
#pragma unroll
        for (int v = 0; v < 8; ++v) {
            float ga = g0[v], gb = g1[v];
            float ha = ga * __builtin_amdgcn_rcpf(1.f + __expf(-ga)) * u0[v];
            float hb = gb * __builtin_amdgcn_rcpf(1.f + __expf(-gb)) * u1[v];
            Hs[(v + hi8) * HPITCH + n0 + nl]      = f2bf(ha);
            Hs[(16 + v + hi8) * HPITCH + n0 + nl] = f2bf(hb);
        }
    }
    __syncthreads();

    // ---- phase B: OUT[32][1024] = (H @ D) * w; each wave owns 128 HIDDEN cols ----
    const __bf16* hrow0 = Hs + nl * HPITCH;
    const __bf16* hrow1 = Hs + (16 + nl) * HPITCH;
    for (int t = 0; t < 8; ++t) {
        const int n0 = wave * 128 + t * 16;
        const __bf16* drow = Dt + (size_t)(n0 + nl) * INTER;
        v8f c0, c1;
#pragma unroll
        for (int i = 0; i < 8; ++i) { c0[i] = 0.f; c1[i] = 0.f; }
        for (int kb = 0; kb < INTER; kb += 32) {
            v16bf a0 = load_a_frag(hrow0, kb, hi8);
            v16bf a1 = load_a_frag(hrow1, kb, hi8);
            v16bf b  = load_b_frag(drow, kb, hi16);
            c0 = wmma_bf16(a0, b, c0);
            c1 = wmma_bf16(a1, b, c1);
        }
#pragma unroll
        for (int v = 0; v < 8; ++v) {
            int m0r = v + hi8, m1r = 16 + v + hi8;
            unsafeAtomicAdd(&out[(size_t)toks[m0r] * HIDDEN + n0 + nl], c0[v] * twgt[m0r]);
            unsafeAtomicAdd(&out[(size_t)toks[m1r] * HIDDEN + n0 + nl], c1[v] * twgt[m1r]);
        }
    }
}

// ---------- host launcher ----------

extern "C" void kernel_launch(void* const* d_in, const int* in_sizes, int n_in,
                              void* d_out, int out_size, void* d_ws, size_t ws_size,
                              hipStream_t stream) {
    const float* x  = (const float*)d_in[0];
    const float* rw = (const float*)d_in[1];
    const float* gw = (const float*)d_in[2];
    const float* uw = (const float*)d_in[3];
    const float* dw = (const float*)d_in[4];
    float* out = (float*)d_out;

    const int T = in_sizes[0] / HIDDEN;          // 16384 tokens
    const int nElem = T * HIDDEN;
    const int n4 = nElem / 4;

    // workspace layout
    char* ws = (char*)d_ws;
    size_t off = 0;
    int*    cnt  = (int*)(ws + off);  off += 256;
    int*    idxb = (int*)(ws + off);  off += (size_t)NUM_EXPERTS * T * 4;
    float*  wtb  = (float*)(ws + off); off += (size_t)NUM_EXPERTS * T * 4;
    off = (off + 255) & ~(size_t)255;
    __bf16* xbf  = (__bf16*)(ws + off); off += (size_t)nElem * 2;
    __bf16* gwt  = (__bf16*)(ws + off); off += (size_t)NUM_EXPERTS * HIDDEN * INTER * 2;
    __bf16* uwt  = (__bf16*)(ws + off); off += (size_t)NUM_EXPERTS * HIDDEN * INTER * 2;
    __bf16* dwt  = (__bf16*)(ws + off); off += (size_t)NUM_EXPERTS * INTER * HIDDEN * 2;

    moe_init_kernel<<<(n4 + 255) / 256, 256, 0, stream>>>(out, cnt, n4);
    moe_cvt_kernel<<<(n4 + 255) / 256, 256, 0, stream>>>(x, xbf, n4);
    // gate/up: [HIDDEN][INTER] -> [INTER][HIDDEN] bf16 ; down: [INTER][HIDDEN] -> [HIDDEN][INTER]
    moe_wt_kernel<<<dim3(INTER / 32, HIDDEN / 32, NUM_EXPERTS), 256, 0, stream>>>(gw, gwt, HIDDEN, INTER);
    moe_wt_kernel<<<dim3(INTER / 32, HIDDEN / 32, NUM_EXPERTS), 256, 0, stream>>>(uw, uwt, HIDDEN, INTER);
    moe_wt_kernel<<<dim3(HIDDEN / 32, INTER / 32, NUM_EXPERTS), 256, 0, stream>>>(dw, dwt, INTER, HIDDEN);
    moe_router_kernel<<<(T + 7) / 8, 256, 0, stream>>>(x, rw, cnt, idxb, wtb, T);

    dim3 grid((T + MTILE - 1) / MTILE, NUM_EXPERTS);
    moe_ffn_kernel<<<grid, 256, SMEM_BYTES, stream>>>(xbf, gwt, uwt, dwt, cnt, idxb, wtb, out, T);
}
